// MoE_4355096838532
// MI455X (gfx1250) — matrix operand order repaired and checked by reading, not compiled, not run on hardware
//
#include <hip/hip_runtime.h>
#include <hip/hip_bf16.h>

// Problem sizes (fixed by the reference)
#define B_  8
#define S_  2048
#define D_  1024
#define E_  4

// GEMM tiling
#define BM  128         // block M tile (rows of x)
#define BN  64          // block N tile (output features)
#define BK  32          // K per stage == WMMA K
#define LDK 56          // padded row length in halfs (112B: 16B-aligned, conflict-free)
#define NK  (D_ / BK)   // 32 k-stages

#ifndef __has_builtin
#define __has_builtin(x) 0
#endif
// TDM builtin: 5-arg on ROCm7.2/clang-22, 6-arg (extra zeroed int32x8 + cpol) on clang-23.
#if __has_builtin(__builtin_amdgcn_tensor_load_to_lds) && \
    __has_builtin(__builtin_amdgcn_s_wait_tensorcnt)
#define USE_TDM 1
#if (__clang_major__ >= 23)
#define TDM_6ARG 1
#else
#define TDM_6ARG 0
#endif
#else
#define USE_TDM 0
#endif

typedef __attribute__((ext_vector_type(16))) __bf16          v16bf;
typedef __attribute__((ext_vector_type(8)))  float           v8f;
typedef __attribute__((ext_vector_type(8)))  unsigned short  v8us;
typedef __attribute__((ext_vector_type(4)))  unsigned short  v4us;

__device__ __forceinline__ unsigned short f2bf(float f) {
    // round-to-nearest-even f32 -> bf16
    unsigned int u = __builtin_bit_cast(unsigned int, f);
    u += 0x7fffu + ((u >> 16) & 1u);
    return (unsigned short)(u >> 16);
}

// Workgroup-relative LDS byte address (low 32 bits of the generic pointer).
__device__ __forceinline__ unsigned lds_addr_of(const void* p) {
    return (unsigned)(unsigned long long)(uintptr_t)p;
}

#if USE_TDM
typedef __attribute__((ext_vector_type(4))) unsigned int v4ui;
typedef __attribute__((ext_vector_type(8))) int          v8i;
typedef __attribute__((ext_vector_type(4))) int          v4i;

// One TDM descriptor: move a (tile0 x tile1 x tile2) bf16 tile (x,y,z) from
// global to LDS, padding each 32-element row (16 dwords) by 12 dwords so the
// LDS row stride is exactly LDK=56 halfs.  strides are in elements (2B).
__device__ __forceinline__ void tdm_tile_to_lds(unsigned lds_byte, const void* gtile,
                                                unsigned tile0, unsigned tile1, unsigned tile2,
                                                unsigned long long stride0,
                                                unsigned long long stride1) {
    const unsigned long long ga = (unsigned long long)(uintptr_t)gtile;
    const unsigned big = 0x40000000u;   // huge tensor dims: never OOB-dropped

    v4ui g0;
    g0[0] = 1u;                                               // count=1 valid user D#
    g0[1] = lds_byte;                                         // lds_addr
    g0[2] = (unsigned)ga;                                     // global_addr[31:0]
    g0[3] = (unsigned)((ga >> 32) & 0x01ffffffu) | (2u << 30);// global_addr[56:32] | type=2

    v8i g1;
    g1[0] = (int)((1u << 16)        // data_size = 2 bytes (bf16)
                | (1u << 20)        // pad_enable
                | (3u << 22)        // pad_interval: 16 dwords  (one 32-half row)
                | (11u << 25));     // pad_amount:   12 dwords  (24 halfs -> stride 56)
    g1[1] = (int)((big & 0xffffu) << 16);                                 // tensor_dim0 lo16
    g1[2] = (int)(((big >> 16) & 0xffffu) | ((big & 0xffffu) << 16));     // dim0 hi | dim1 lo
    g1[3] = (int)(((big >> 16) & 0xffffu) | ((tile0 & 0xffffu) << 16));   // dim1 hi | tile_dim0
    g1[4] = (int)((tile1 & 0xffffu) | ((tile2 & 0xffffu) << 16));         // tile_dim1 | tile_dim2
    g1[5] = (int)(unsigned)(stride0 & 0xffffffffu);                       // dim0_stride lo32
    g1[6] = (int)((unsigned)((stride0 >> 32) & 0xffffu) |
                  ((unsigned)(stride1 & 0xffffu) << 16));                 // dim0_str hi | dim1_str lo16
    g1[7] = (int)(unsigned)((stride1 >> 16) & 0xffffffffu);               // dim1_stride[47:16]

    v4i g2;
    g2[0] = (int)big;  g2[1] = 0;  g2[2] = 0;  g2[3] = 0;                 // tensor_dim2 | no dim3
    v4i g3;
    g3[0] = 0;  g3[1] = 0;  g3[2] = 0;  g3[3] = 0;

#if TDM_6ARG
    const v8i g4 = {0, 0, 0, 0, 0, 0, 0, 0};                              // VADDR4: unused
    __builtin_amdgcn_tensor_load_to_lds(g0, g1, g2, g3, g4, 0);
#else
    __builtin_amdgcn_tensor_load_to_lds(g0, g1, g2, g3, 0);
#endif
}
#endif

// ---------------------------------------------------------------------------
// Kernel 0: zero the per-(b,e) weight accumulator in workspace
// ---------------------------------------------------------------------------
__global__ void moe_zero_w(float* __restrict__ w) {
    if (threadIdx.x < B_ * E_) w[threadIdx.x] = 0.0f;
}

// ---------------------------------------------------------------------------
// Kernel 0b: bulk f32 -> bf16 conversion (vectorized, memory-bound)
// ---------------------------------------------------------------------------
__global__ __launch_bounds__(256) void f32_to_bf16(const float* __restrict__ src,
                                                   unsigned short* __restrict__ dst,
                                                   int n4) {
    const int i = blockIdx.x * 256 + threadIdx.x;
    if (i < n4) {
        const float4 v = ((const float4*)src)[i];
        ((v4us*)dst)[i] = (v4us){f2bf(v.x), f2bf(v.y), f2bf(v.z), f2bf(v.w)};
    }
}

// ---------------------------------------------------------------------------
// Kernel 1: gating -> weights[b,e] = count(argmax==e)/S
// ---------------------------------------------------------------------------
__global__ __launch_bounds__(256) void moe_gate(const float* __restrict__ x,
                                                const float* __restrict__ Wg,
                                                const float* __restrict__ bg,
                                                float* __restrict__ wgt) {
    __shared__ float sWg[E_ * D_];
    const int tid = threadIdx.x;
    for (int i = tid; i < E_ * D_; i += 256) sWg[i] = Wg[i];
    __syncthreads();

    const int g = blockIdx.x * 256 + tid;      // [0, B*S)
    const int b = g / S_;
    const int s = g % S_;

    float acc[E_];
#pragma unroll
    for (int e = 0; e < E_; ++e) acc[e] = bg[e];

    const float4* xv = (const float4*)(x + ((size_t)b * S_ + s) * D_);
#pragma unroll 4
    for (int dv = 0; dv < D_ / 4; ++dv) {
        const float4 v = xv[dv];
#pragma unroll
        for (int e = 0; e < E_; ++e) {
            const float4 gv = ((const float4*)(sWg + e * D_))[dv];
            acc[e] += v.x * gv.x + v.y * gv.y + v.z * gv.z + v.w * gv.w;
        }
    }
    int best = 0;
#pragma unroll
    for (int e = 1; e < E_; ++e) if (acc[e] > acc[best]) best = e;  // first-max, like jnp.argmax
    atomicAdd(&wgt[b * E_ + best], 1.0f / (float)S_);               // exact: 1/2048 is a power of two
}

// ---------------------------------------------------------------------------
// Kernel 2: fused 4-expert GEMM + bias + ReLU + weighted combine.
//   out[b,m,n] = sum_e w[b,e] * max( sum_k x[b,m,k]*We[e,n,k] + be[e,n], 0 )
// Block: 256 threads (8 wave32), tile 128x64; wave tile 64x16, 4 experts.
// MODE 0: f32 inputs, convert while staging (fallback, small workspace)
// MODE 1: bf16 inputs (preconverted), manual global->reg->LDS staging
// MODE 2: bf16 inputs, Tensor Data Mover global->LDS DMA (2 TDM ops/stage)
// ---------------------------------------------------------------------------
template <int MODE>
__global__ __launch_bounds__(256) void moe_expert_gemm(const void* __restrict__ xsrc,
                                                       const void* __restrict__ wesrc,
                                                       const float* __restrict__ be,
                                                       const float* __restrict__ wgt,
                                                       float* __restrict__ out) {
    __shared__ unsigned short As[2][BM * LDK];          // 28 KB  (x tile, bf16)
    __shared__ unsigned short Bs[2][E_ * BN * LDK];     // 56 KB  (We tiles, bf16, 4 experts)

    const int tid  = threadIdx.x;
    const int lane = tid & 31;
    const int wave = tid >> 5;
    const int wm   = wave & 1;                 // M half   (2 x 64 rows)
    const int wn   = wave >> 1;                // N group  (4 x 16 cols)
    const int b    = blockIdx.z;
    const int m0   = blockIdx.y * BM;
    const int n0   = blockIdx.x * BN;

    const float*          xf  = (const float*)xsrc;
    const float*          wef = (const float*)wesrc;
    const unsigned short* xh  = (const unsigned short*)xsrc;
    const unsigned short* weh = (const unsigned short*)wesrc;
    const size_t          xoff = (size_t)b * S_ * D_;

    const v8f z8 = {0.f, 0.f, 0.f, 0.f, 0.f, 0.f, 0.f, 0.f};
    v8f acc[E_][4];
#pragma unroll
    for (int e = 0; e < E_; ++e)
#pragma unroll
        for (int mf = 0; mf < 4; ++mf) acc[e][mf] = z8;

    // staging registers (MODE 0/1 only; dead for MODE 2)
    v8us aP[2]; v8us bP[4];          // MODE 1 (bf16 direct)
    v4us aC[4]; v4us bC[8];          // MODE 0 (convert)

    auto loadStage = [&](int kt) {
        if (MODE == 1) {
#pragma unroll
            for (int i = 0; i < 2; ++i) {                 // 128 rows x 4 x v8us
                const int idx = tid + 256 * i;
                const int row = idx >> 2, vec = idx & 3;
                aP[i] = *(const v8us*)(xh + xoff + (size_t)(m0 + row) * D_ + kt * BK + vec * 8);
            }
#pragma unroll
            for (int i = 0; i < 4; ++i) {                 // 4e x 64 rows x 4 x v8us
                const int idx = tid + 256 * i;
                const int vec = idx & 3, row = (idx >> 2) & 63, e = idx >> 8;
                bP[i] = *(const v8us*)(weh + ((size_t)e * D_ + n0 + row) * D_ + kt * BK + vec * 8);
            }
        } else if (MODE == 0) {
#pragma unroll
            for (int i = 0; i < 4; ++i) {                 // 128 rows x 8 float4
                const int idx = tid + 256 * i;
                const int row = idx >> 3, vec = idx & 7;
                const float4 v = *(const float4*)(xf + xoff + (size_t)(m0 + row) * D_ + kt * BK + vec * 4);
                aC[i] = (v4us){f2bf(v.x), f2bf(v.y), f2bf(v.z), f2bf(v.w)};
            }
#pragma unroll
            for (int i = 0; i < 8; ++i) {                 // 4e x 64 rows x 8 float4
                const int idx = tid + 256 * i;
                const int vec = idx & 7, row = (idx >> 3) & 63, e = idx >> 9;
                const float4 v = *(const float4*)(wef + ((size_t)e * D_ + n0 + row) * D_ + kt * BK + vec * 4);
                bC[i] = (v4us){f2bf(v.x), f2bf(v.y), f2bf(v.z), f2bf(v.w)};
            }
        }
    };
    auto storeStage = [&](int buf) {
        if (MODE == 1) {
#pragma unroll
            for (int i = 0; i < 2; ++i) {
                const int idx = tid + 256 * i;
                const int row = idx >> 2, vec = idx & 3;
                *(v8us*)&As[buf][row * LDK + vec * 8] = aP[i];
            }
#pragma unroll
            for (int i = 0; i < 4; ++i) {
                const int idx = tid + 256 * i;
                const int vec = idx & 3, row = (idx >> 2) & 63, e = idx >> 8;
                *(v8us*)&Bs[buf][(e * BN + row) * LDK + vec * 8] = bP[i];
            }
        } else if (MODE == 0) {
#pragma unroll
            for (int i = 0; i < 4; ++i) {
                const int idx = tid + 256 * i;
                const int row = idx >> 3, vec = idx & 7;
                *(v4us*)&As[buf][row * LDK + vec * 4] = aC[i];
            }
#pragma unroll
            for (int i = 0; i < 8; ++i) {
                const int idx = tid + 256 * i;
                const int vec = idx & 7, row = (idx >> 3) & 63, e = idx >> 9;
                *(v4us*)&Bs[buf][(e * BN + row) * LDK + vec * 4] = bC[i];
            }
        }
    };

#if USE_TDM
    auto tdmStage = [&](int buf, int kt) {
        // A tile: 2D, 32(K) x 128(rows), row stride D elements
        tdm_tile_to_lds(lds_addr_of(&As[buf][0]),
                        xh + xoff + (size_t)m0 * D_ + kt * BK,
                        BK, BM, 0, (unsigned long long)D_, 0ull);
        // B tiles: 3D, 32(K) x 64(rows) x 4(experts): one descriptor for all experts
        tdm_tile_to_lds(lds_addr_of(&Bs[buf][0]),
                        weh + (size_t)n0 * D_ + kt * BK,
                        BK, BN, E_, (unsigned long long)D_,
                        (unsigned long long)D_ * (unsigned long long)D_);
    };
    if (MODE == 2) {
        if (wave == 0) {
            tdmStage(0, 0);
            __builtin_amdgcn_s_wait_tensorcnt(0);
        }
    }
#endif
    if (MODE != 2) {
        loadStage(0);
        storeStage(0);
    }
    __syncthreads();

    int buf = 0;
    for (int kt = 0; kt < NK; ++kt) {
#if USE_TDM
        if (MODE == 2) {
            if (wave == 0 && kt + 1 < NK) tdmStage(buf ^ 1, kt + 1);
        }
#endif
        if (MODE != 2 && kt + 1 < NK) loadStage(kt + 1);
        if (kt + 2 < NK) {
            // global_prefetch_b8 for the k+2 tiles
            if (MODE != 0) {
                __builtin_prefetch(xh + xoff + (size_t)(m0 + (tid & 127)) * D_ + (kt + 2) * BK, 0, 1);
                __builtin_prefetch(weh + ((size_t)(tid >> 6) * D_ + n0 + (tid & 63)) * D_ + (kt + 2) * BK, 0, 1);
            } else {
                __builtin_prefetch(xf + xoff + (size_t)(m0 + (tid >> 1)) * D_ + (kt + 2) * BK + (tid & 1) * 16, 0, 1);
                __builtin_prefetch(wef + ((size_t)(tid >> 6) * D_ + n0 + (tid & 63)) * D_ + (kt + 2) * BK, 0, 1);
            }
        }

        // ---- A fragments (16x32 bf16 per ISA 7.12.2) ----
        v16bf afrag[4];
        {
            const int koff = (lane >> 4) * 8;
#pragma unroll
            for (int mf = 0; mf < 4; ++mf) {
                const int r = wm * 64 + mf * 16 + (lane & 15);
                const v8us lo = *(const v8us*)&As[buf][r * LDK + koff];
                const v8us hi = *(const v8us*)&As[buf][r * LDK + 16 + koff];
                afrag[mf] = __builtin_bit_cast(
                    v16bf, __builtin_shufflevector(lo, hi, 0, 1, 2, 3, 4, 5, 6, 7,
                                                   8, 9, 10, 11, 12, 13, 14, 15));
            }
        }
        // ---- B fragment per expert (32x16 bf16), 4 wmma each ----
#pragma unroll
        for (int e = 0; e < E_; ++e) {
            const int c = wn * 16 + (lane & 15);
            const int k = (lane >> 4) * 8;
            const unsigned short* bp = &Bs[buf][(e * BN + c) * LDK + k];
            const v8us lo = *(const v8us*)bp;
            const v8us hi = *(const v8us*)(bp + 16);
            const v16bf bfrag = __builtin_bit_cast(
                v16bf, __builtin_shufflevector(lo, hi, 0, 1, 2, 3, 4, 5, 6, 7,
                                               8, 9, 10, 11, 12, 13, 14, 15));
#pragma unroll
            for (int mf = 0; mf < 4; ++mf) {
                acc[e][mf] = __builtin_amdgcn_wmma_f32_16x16x32_bf16(
                    false, afrag[mf], false, bfrag, (short)0, acc[e][mf], false, false);
            }
        }

        if (MODE != 2 && kt + 1 < NK) storeStage(buf ^ 1);
#if USE_TDM
        if (MODE == 2) {
            if (wave == 0 && kt + 1 < NK) __builtin_amdgcn_s_wait_tensorcnt(0);
        }
#endif
        __syncthreads();
        buf ^= 1;
    }

    // ---- epilogue: bias + ReLU per expert, weighted combine, store ----
    float w[E_];
#pragma unroll
    for (int e = 0; e < E_; ++e) w[e] = wgt[b * E_ + e];

    const int c = n0 + wn * 16 + (lane & 15);
    float bias[E_];
#pragma unroll
    for (int e = 0; e < E_; ++e) bias[e] = be[e * D_ + c];
    const int rbase = m0 + wm * 64 + ((lane >> 4) ? 8 : 0);

#pragma unroll
    for (int mf = 0; mf < 4; ++mf) {
#pragma unroll
        for (int v = 0; v < 8; ++v) {
            float o = 0.0f;
#pragma unroll
            for (int e = 0; e < E_; ++e) {
                const float h = acc[e][mf][v] + bias[e];
                o += w[e] * (h > 0.0f ? h : 0.0f);
            }
            out[((size_t)b * S_ + rbase + mf * 16 + v) * D_ + c] = o;
        }
    }
}

// ---------------------------------------------------------------------------
extern "C" void kernel_launch(void* const* d_in, const int* in_sizes, int n_in,
                              void* d_out, int out_size, void* d_ws, size_t ws_size,
                              hipStream_t stream) {
    (void)in_sizes; (void)n_in; (void)out_size;
    const float* x  = (const float*)d_in[0];   // [B,S,D]
    const float* Wg = (const float*)d_in[1];   // [E,D]
    const float* bg = (const float*)d_in[2];   // [E]
    const float* We = (const float*)d_in[3];   // [E,D,D]
    const float* be = (const float*)d_in[4];   // [E,D]
    float* out = (float*)d_out;                // [B,S,D]

    float* wgt = (float*)d_ws;                 // [B,E] expert weights at ws[0:128)
    const size_t nx = (size_t)B_ * S_ * D_;    // 16.78M elems
    const size_t nw = (size_t)E_ * D_ * D_;    // 4.19M  elems
    const size_t need = 256 + (nx + nw) * sizeof(unsigned short);
    const bool   pre  = ws_size >= need;

    moe_zero_w<<<1, 64, 0, stream>>>(wgt);
    moe_gate<<<(B_ * S_) / 256, 256, 0, stream>>>(x, Wg, bg, wgt);

    dim3 grid(D_ / BN, S_ / BM, B_);
    if (pre) {
        unsigned short* xbf  = (unsigned short*)((char*)d_ws + 256);
        unsigned short* webf = xbf + nx;
        f32_to_bf16<<<(int)(nx / 4 + 255) / 256, 256, 0, stream>>>(x, xbf, (int)(nx / 4));
        f32_to_bf16<<<(int)(nw / 4 + 255) / 256, 256, 0, stream>>>(We, webf, (int)(nw / 4));
#if USE_TDM
        moe_expert_gemm<2><<<grid, 256, 0, stream>>>((const void*)xbf, (const void*)webf,
                                                     be, wgt, out);
#else
        moe_expert_gemm<1><<<grid, 256, 0, stream>>>((const void*)xbf, (const void*)webf,
                                                     be, wgt, out);
#endif
    } else {
        moe_expert_gemm<0><<<grid, 256, 0, stream>>>((const void*)x, (const void*)We,
                                                     be, wgt, out);
    }
}
